// Model_15006615734384
// MI455X (gfx1250) — compile-verified, matrix operand order checked by reading, not run on hardware
//
#include <hip/hip_runtime.h>
#include <hip/hip_bf16.h>
#include <stdint.h>

// Problem constants (from reference)
#define BATCH  1024
#define SLEV   60
#define TSEQ   62        // S + 2
#define CIN    25
#define DMODEL 256
#define HID    256
#define G4     1024      // 4*H gate width
#define H2     512       // 2*H bidirectional concat

typedef __attribute__((ext_vector_type(16))) __bf16 v16bf;
typedef __attribute__((ext_vector_type(8)))  __bf16 v8bf;
typedef __attribute__((ext_vector_type(8)))  float  v8f;

static __device__ __forceinline__ float sigf(float x)    { return 1.0f / (1.0f + __expf(-x)); }
static __device__ __forceinline__ float tanhfast(float x){ return 2.0f * sigf(2.0f * x) - 1.0f; }

// Wave-relative LDS byte offset of a __shared__ pointer (low 32 bits of the
// flat address are the group-segment offset on gfx12xx).
static __device__ __forceinline__ unsigned lds_off(const void* p) {
  return (unsigned)(uintptr_t)p;
}

// Async 16B copy global -> LDS (GLOBAL_LOAD_ASYNC_TO_LDS_B128, ASYNCcnt).
static __device__ __forceinline__ void async_g2l_b128(const void* gsrc, const void* ldst) {
  asm volatile("global_load_async_to_lds_b128 %0, %1, off"
               :: "v"(lds_off(ldst)), "v"(gsrc) : "memory");
}
// Async 16B copy LDS -> global (GLOBAL_STORE_ASYNC_FROM_LDS_B128, ASYNCcnt).
static __device__ __forceinline__ void async_l2g_b128(void* gdst, const void* lsrc) {
  asm volatile("global_store_async_from_lds_b128 %0, %1, off"
               :: "v"(gdst), "v"(lds_off(lsrc)) : "memory");
}
static __device__ __forceinline__ void wait_async0() {
  asm volatile("s_wait_asynccnt 0" ::: "memory");
}

// Load one 16x32 bf16 WMMA operand tile.
// ISA 16-bit A layout (CDNA5 ISA §7.12.2): lanes 0-15 hold row M=lane,
// K={k0..k0+7} in v0..3 and K={k0+16..k0+23} in v4..7; lanes 16-31 hold the
// K+8 shifted halves. B operand (N x K weight rows) uses the mirrored rule.
static __device__ __forceinline__ v16bf load_frag(const __bf16* base, int rowstride, int lane) {
  const int half = lane >> 4;
  const __bf16* r = base + (lane & 15) * rowstride + 8 * half;
  v8bf lo = *(const v8bf*)(r);        // K = k0 + 8*half      .. +8
  v8bf hi = *(const v8bf*)(r + 16);   // K = k0 + 16 + 8*half .. +8
  return __builtin_shufflevector(lo, hi, 0,1,2,3,4,5,6,7,8,9,10,11,12,13,14,15);
}

// ---------------------------------------------------------------------------
// Fused LSTM scan: one block = 32 batch rows for all T steps of one direction.
// 8 waves; wave w owns hidden dims [32w, 32w+32) across all four gates, so the
// gate nonlinearity is entirely lane-local. h state lives in LDS (16KB),
// c state lives in VGPRs. x_t tile is async-staged into LDS once per step;
// weights stream from L2 as WMMA B-fragments.
// ---------------------------------------------------------------------------
template<int DIN>
__global__ __launch_bounds__(256)
void lstm_scan(const __bf16* __restrict__ xin,   // (T, B, DIN) time-major bf16
               const __bf16* __restrict__ Wih,   // (2, G4, DIN) bf16
               const __bf16* __restrict__ Whh,   // (2, G4, HID) bf16
               const float*  __restrict__ bsum,  // (2, G4) = bih + bhh
               __bf16*       __restrict__ hout)  // (T, B, H2); dir slice [dir*HID, +HID)
{
  __shared__ __align__(16) __bf16 hLDS[32 * HID];   // 16 KB current h tile
  __shared__ __align__(16) __bf16 xLDS[32 * DIN];   // 16/32 KB x_t tile
  const int  tid   = threadIdx.x;
  const int  wave  = tid >> 5;          // 0..7
  const int  lane  = tid & 31;
  const int  dir   = blockIdx.y;        // 0 = forward, 1 = backward
  const int  bbase = blockIdx.x * 32;
  const bool rev   = (dir != 0);

  const __bf16* Wi = Wih + (size_t)dir * G4 * DIN;
  const __bf16* Wh = Whh + (size_t)dir * G4 * HID;
  const float*  bs = bsum + (size_t)dir * G4;

  // Per-lane gate bias, constant over time: N = 256*g + 32*wave + 16*s + lane%16
  float bias[4][2];
  #pragma unroll
  for (int g = 0; g < 4; ++g)
    #pragma unroll
    for (int s = 0; s < 2; ++s)
      bias[g][s] = bs[256 * g + 32 * wave + 16 * s + (lane & 15)];

  // c state in VGPRs: c[mt][s] rows M=16mt+r+8*(lane>>4), col j=32w+16s+lane%16
  v8f c[2][2];
  #pragma unroll
  for (int mt = 0; mt < 2; ++mt)
    #pragma unroll
    for (int s = 0; s < 2; ++s)
      #pragma unroll
      for (int r = 0; r < 8; ++r) c[mt][s][r] = 0.0f;

  for (int i = tid; i < 32 * HID; i += 256) hLDS[i] = (__bf16)0.0f;

  constexpr int XCHUNKS = 32 * DIN / 8;   // 16B chunks in the x tile
  constexpr int PER_THR = XCHUNKS / 256;  // 4 (layer0) or 8 (layer1)

  for (int tt = 0; tt < TSEQ; ++tt) {
    const int t = rev ? (TSEQ - 1 - tt) : tt;

    // ---- async-stage the contiguous 32xDIN activation tile into LDS
    const __bf16* xrow = xin + ((size_t)t * BATCH + bbase) * DIN;
    #pragma unroll
    for (int i = 0; i < PER_THR; ++i) {
      const int idx = tid + i * 256;                 // 16B chunk index
      async_g2l_b128(xrow + idx * 8, xLDS + idx * 8);
    }
    // Prefetch next timestep's tile into L2 while we compute.
    if (tt + 1 < TSEQ) {
      const int tn = rev ? (TSEQ - 2 - tt) : (tt + 1);
      __builtin_prefetch(xin + ((size_t)tn * BATCH + bbase) * DIN + tid * (32 * DIN / 256), 0, 0);
    }
    wait_async0();      // staging (and last step's async store-out) complete
    __syncthreads();    // x tile + h tile visible to all waves

    // Accumulators seeded with bias (bias is per-N, identical across M rows).
    v8f acc[2][4][2];
    #pragma unroll
    for (int mt = 0; mt < 2; ++mt)
      #pragma unroll
      for (int g = 0; g < 4; ++g)
        #pragma unroll
        for (int s = 0; s < 2; ++s)
          #pragma unroll
          for (int r = 0; r < 8; ++r) acc[mt][g][s][r] = bias[g][s];

    // ---- input contribution: g += x_t @ Wih^T  (A from LDS)
    #pragma unroll 4
    for (int k0 = 0; k0 < DIN; k0 += 32) {
      v16bf a0 = load_frag(xLDS + k0,            DIN, lane);
      v16bf a1 = load_frag(xLDS + 16 * DIN + k0, DIN, lane);
      #pragma unroll
      for (int g = 0; g < 4; ++g)
        #pragma unroll
        for (int s = 0; s < 2; ++s) {
          const int n0 = 256 * g + 32 * wave + 16 * s;
          v16bf bfrag = load_frag(Wi + (size_t)n0 * DIN + k0, DIN, lane);
          acc[0][g][s] = __builtin_amdgcn_wmma_f32_16x16x32_bf16(
              false, a0, false, bfrag, (short)0, acc[0][g][s], false, false);
          acc[1][g][s] = __builtin_amdgcn_wmma_f32_16x16x32_bf16(
              false, a1, false, bfrag, (short)0, acc[1][g][s], false, false);
        }
    }

    // ---- recurrent contribution: g += h_{t-1} @ Whh^T  (A from LDS)
    #pragma unroll 4
    for (int k0 = 0; k0 < HID; k0 += 32) {
      v16bf a0 = load_frag(hLDS + k0,            HID, lane);
      v16bf a1 = load_frag(hLDS + 16 * HID + k0, HID, lane);
      #pragma unroll
      for (int g = 0; g < 4; ++g)
        #pragma unroll
        for (int s = 0; s < 2; ++s) {
          const int n0 = 256 * g + 32 * wave + 16 * s;
          v16bf bfrag = load_frag(Wh + (size_t)n0 * HID + k0, HID, lane);
          acc[0][g][s] = __builtin_amdgcn_wmma_f32_16x16x32_bf16(
              false, a0, false, bfrag, (short)0, acc[0][g][s], false, false);
          acc[1][g][s] = __builtin_amdgcn_wmma_f32_16x16x32_bf16(
              false, a1, false, bfrag, (short)0, acc[1][g][s], false, false);
        }
    }

    __syncthreads();   // everyone is done reading hLDS/xLDS before overwrite

    // ---- lane-local LSTM cell: i,f,g,o share (lane, vgpr-row) across tiles
    #pragma unroll
    for (int mt = 0; mt < 2; ++mt)
      #pragma unroll
      for (int s = 0; s < 2; ++s)
        #pragma unroll
        for (int r = 0; r < 8; ++r) {
          const float iv = acc[mt][0][s][r];
          const float fv = acc[mt][1][s][r];
          const float gv = acc[mt][2][s][r];
          const float ov = acc[mt][3][s][r];
          const float cc = sigf(fv) * c[mt][s][r] + sigf(iv) * tanhfast(gv);
          c[mt][s][r] = cc;
          const float hh = sigf(ov) * tanhfast(cc);
          const int M = 16 * mt + r + 8 * (lane >> 4);
          const int j = 32 * wave + 16 * s + (lane & 15);
          hLDS[M * HID + j] = (__bf16)hh;
        }

    __syncthreads();   // new h visible to all waves

    // ---- async store-out of the h tile (LDS -> global, no VGPR round-trip).
    // Completion is guaranteed by next step's wait_async0 (before hLDS is
    // overwritten) and by S_ENDPGM's implicit wait-idle on the last step.
    __bf16* orow = hout + ((size_t)t * BATCH + bbase) * H2 + dir * HID;
    #pragma unroll
    for (int it = 0; it < 4; ++it) {
      const int idx = tid + it * 256;   // 0..1023
      const int row = idx >> 5;         // 0..31 batch rows
      const int ch  = idx & 31;         // 16B chunks of the 512B row
      async_l2g_b128(orow + (size_t)row * H2 + ch * 8, hLDS + row * HID + ch * 8);
    }
  }
}

// ---------------------------------------------------------------------------
// Embedding: seq[0]=seq[61]=gx, seq[1..60]=lx, stored bf16 time-major (T,B,D).
// One block per batch row; d = threadIdx.x; x row loads are wave-uniform.
// ---------------------------------------------------------------------------
__global__ void embed_kernel(const float* __restrict__ x,
                             const float* __restrict__ Wg, const float* __restrict__ bg,
                             const float* __restrict__ Wl, const float* __restrict__ bl,
                             __bf16* __restrict__ seq)
{
  const int b = blockIdx.x;
  const int d = threadIdx.x;                 // 0..255
  const float* xb = x + (size_t)b * SLEV * CIN;

  float g = bg[d];
  #pragma unroll
  for (int k = 0; k < 16; ++k) g += xb[9 + k] * Wg[d * 16 + k];
  const __bf16 gb = (__bf16)g;
  seq[((size_t)0 * BATCH + b) * DMODEL + d]           = gb;
  seq[((size_t)(TSEQ - 1) * BATCH + b) * DMODEL + d]  = gb;

  for (int t = 0; t < SLEV; ++t) {
    float v = bl[d];
    #pragma unroll
    for (int k = 0; k < 9; ++k) v += xb[t * CIN + k] * Wl[d * 9 + k];
    seq[((size_t)(t + 1) * BATCH + b) * DMODEL + d] = (__bf16)v;
  }
}

__global__ void cvt_bf16(const float* __restrict__ src, __bf16* __restrict__ dst, int n) {
  int i = blockIdx.x * blockDim.x + threadIdx.x;
  const int stride = gridDim.x * blockDim.x;
  for (; i < n; i += stride) dst[i] = (__bf16)src[i];
}

__global__ void bias_sum(const float* __restrict__ a, const float* __restrict__ b,
                         float* __restrict__ dst, int n) {
  const int i = blockIdx.x * blockDim.x + threadIdx.x;
  if (i < n) dst[i] = a[i] + b[i];
}

// gout[b] = concat(h1[t=0,b], h1[t=61,b]) @ Wgp^T + bgp   -> (B, 8)
__global__ void gout_kernel(const __bf16* __restrict__ h1, const float* __restrict__ Wgp,
                            const float* __restrict__ bgp, float* __restrict__ gout)
{
  const int b = blockIdx.x * blockDim.x + threadIdx.x;
  if (b >= BATCH) return;
  const __bf16* h0r = h1 + ((size_t)0 * BATCH + b) * H2;
  const __bf16* hTr = h1 + ((size_t)(TSEQ - 1) * BATCH + b) * H2;
  float acc[8];
  #pragma unroll
  for (int o = 0; o < 8; ++o) acc[o] = bgp[o];
  for (int j = 0; j < H2; ++j) {
    const float a = (float)h0r[j];
    const float z = (float)hTr[j];
    #pragma unroll
    for (int o = 0; o < 8; ++o)
      acc[o] += a * Wgp[o * 1024 + j] + z * Wgp[o * 1024 + 512 + j];
  }
  #pragma unroll
  for (int o = 0; o < 8; ++o) gout[b * 8 + o] = acc[o];
}

// out[b,s,0:6] = h1[s+1,b] @ Wlp^T + blp ; out[b,s,6:14] = gout[b]
__global__ void final_kernel(const __bf16* __restrict__ h1, const float* __restrict__ Wlp,
                             const float* __restrict__ blp, const float* __restrict__ gout,
                             float* __restrict__ out)
{
  const int idx = blockIdx.x * blockDim.x + threadIdx.x;   // b*SLEV + s
  if (idx >= BATCH * SLEV) return;
  const int b = idx / SLEV, s = idx % SLEV;
  const __bf16* hr = h1 + ((size_t)(s + 1) * BATCH + b) * H2;
  float acc[6];
  #pragma unroll
  for (int o = 0; o < 6; ++o) acc[o] = blp[o];
  for (int j = 0; j < H2; ++j) {
    const float v = (float)hr[j];
    #pragma unroll
    for (int o = 0; o < 6; ++o) acc[o] += v * Wlp[o * H2 + j];
  }
  float* orow = out + (size_t)idx * 14;
  #pragma unroll
  for (int o = 0; o < 6; ++o) orow[o] = acc[o];
  #pragma unroll
  for (int o = 0; o < 8; ++o) orow[6 + o] = gout[b * 8 + o];
}

// ---------------------------------------------------------------------------
extern "C" void kernel_launch(void* const* d_in, const int* in_sizes, int n_in,
                              void* d_out, int out_size, void* d_ws, size_t ws_size,
                              hipStream_t stream)
{
  (void)in_sizes; (void)n_in; (void)out_size; (void)ws_size;
  const float* x    = (const float*)d_in[0];
  const float* Wg   = (const float*)d_in[1];
  const float* bg   = (const float*)d_in[2];
  const float* Wl   = (const float*)d_in[3];
  const float* bl   = (const float*)d_in[4];
  const float* Wih0 = (const float*)d_in[5];
  const float* Whh0 = (const float*)d_in[6];
  const float* bih0 = (const float*)d_in[7];
  const float* bhh0 = (const float*)d_in[8];
  const float* Wih1 = (const float*)d_in[9];
  const float* Whh1 = (const float*)d_in[10];
  const float* bih1 = (const float*)d_in[11];
  const float* bhh1 = (const float*)d_in[12];
  const float* Wgp  = (const float*)d_in[13];
  const float* bgp  = (const float*)d_in[14];
  const float* Wlp  = (const float*)d_in[15];
  const float* blp  = (const float*)d_in[16];
  float* out = (float*)d_out;

  // Workspace carve-up (all sizes 256B-aligned). ~168 MB total, L2 resident.
  char* ws = (char*)d_ws;
  size_t off = 0;
  __bf16* seq   = (__bf16*)(ws + off); off += (size_t)TSEQ * BATCH * DMODEL * 2;
  __bf16* h0    = (__bf16*)(ws + off); off += (size_t)TSEQ * BATCH * H2 * 2;
  __bf16* h1    = (__bf16*)(ws + off); off += (size_t)TSEQ * BATCH * H2 * 2;
  __bf16* wih0b = (__bf16*)(ws + off); off += (size_t)2 * G4 * DMODEL * 2;
  __bf16* whh0b = (__bf16*)(ws + off); off += (size_t)2 * G4 * HID * 2;
  __bf16* wih1b = (__bf16*)(ws + off); off += (size_t)2 * G4 * H2 * 2;
  __bf16* whh1b = (__bf16*)(ws + off); off += (size_t)2 * G4 * HID * 2;
  float*  bsum0 = (float*)(ws + off);  off += (size_t)2 * G4 * 4;
  float*  bsum1 = (float*)(ws + off);  off += (size_t)2 * G4 * 4;
  float*  goutb = (float*)(ws + off);  off += (size_t)BATCH * 8 * 4;

  // Preprocessing: weight conversion to bf16 + bias folding + embeddings.
  cvt_bf16<<<256, 256, 0, stream>>>(Wih0, wih0b, 2 * G4 * DMODEL);
  cvt_bf16<<<256, 256, 0, stream>>>(Whh0, whh0b, 2 * G4 * HID);
  cvt_bf16<<<256, 256, 0, stream>>>(Wih1, wih1b, 2 * G4 * H2);
  cvt_bf16<<<256, 256, 0, stream>>>(Whh1, whh1b, 2 * G4 * HID);
  bias_sum<<<8, 256, 0, stream>>>(bih0, bhh0, bsum0, 2 * G4);
  bias_sum<<<8, 256, 0, stream>>>(bih1, bhh1, bsum1, 2 * G4);
  embed_kernel<<<BATCH, DMODEL, 0, stream>>>(x, Wg, bg, Wl, bl, seq);

  // Fused WMMA scans: 32 batch tiles x 2 directions per layer.
  dim3 sgrid(BATCH / 32, 2);
  lstm_scan<DMODEL><<<sgrid, 256, 0, stream>>>(seq, wih0b, whh0b, bsum0, h0);
  lstm_scan<H2>    <<<sgrid, 256, 0, stream>>>(h0,  wih1b, whh1b, bsum1, h1);

  // Heads.
  gout_kernel<<<BATCH / 256, 256, 0, stream>>>(h1, Wgp, bgp, goutb);
  final_kernel<<<(BATCH * SLEV + 255) / 256, 256, 0, stream>>>(h1, Wlp, blp, goutb, out);
}
